// CustomLSTMPeephole_59528246722719
// MI455X (gfx1250) — matrix operand order repaired and checked by reading, not compiled, not run on hardware
//
#include <hip/hip_runtime.h>
#include <hip/hip_bf16.h>

// Problem constants (match reference): B=64, T=512, H=512, 4H=2048
#define B_SZ 64
#define T_SZ 512
#define H_SZ 512
#define G_SZ 2048

typedef __attribute__((ext_vector_type(16))) __bf16 v16bf;
typedef __attribute__((ext_vector_type(8)))  __bf16 v8bf;
typedef __attribute__((ext_vector_type(8)))  float  v8f;
typedef __attribute__((ext_vector_type(4)))  float  v4f;

// ---------------------------------------------------------------------------
// Pack a fp32 weight matrix [H x 4H] (row-major) into bf16 transposed [4H x H]
// so that WMMA B-fragments (column n, K contiguous) are contiguous 32B loads.
// ---------------------------------------------------------------------------
__global__ void pack_transpose_bf16(const float* __restrict__ in,
                                    __bf16* __restrict__ out) {
    int idx = blockIdx.x * blockDim.x + threadIdx.x;   // over 4H*H
    int g = idx >> 9;            // output row (gate column index), /H
    int k = idx & (H_SZ - 1);    // output col (K index)
    out[idx] = (__bf16)in[(size_t)k * G_SZ + g];
}

// Initialize ping-pong cell-state buffers (fp32 + bf16 copies) from c0.
__global__ void init_c(const float* __restrict__ c0,
                       float* __restrict__ cf, __bf16* __restrict__ cb) {
    int idx = blockIdx.x * blockDim.x + threadIdx.x;   // over B*H
    float v = c0[idx];
    cf[idx] = v;
    cb[idx] = (__bf16)v;
}

// ---------------------------------------------------------------------------
// xu[(t*B + b)*4H + g] = x[b,t,:] . U[:,g] + bias[g]
// M = B*T = 32768, N = 4H = 2048, K = H = 512.
// Block: 128 threads = 4 waves, macro-tile 128(M) x 128(N).
// Each wave owns a 32(M) x 128(N) strip: 2 A-fragments x 8 B-subtiles = 16
// accumulators, so every LDS B-fragment feeds TWO WMMAs (2x matrix-op work
// per ds_load vs the 1:2 ratio before).
// The 8KB B-tile (128 cols x 32 K, bf16) is shared by all 4 waves: staged
// into LDS with gfx1250 async-to-LDS copies, double-buffered on ASYNCcnt, so
// B L2 latency overlaps the 16 WMMAs of the previous K-slice.
// A (x, fp32) is converted to bf16 on the fly (per-wave unique rows).
// ---------------------------------------------------------------------------
#define BCOL_STRIDE 40   // padded column stride in elements (80B: 16B-aligned,
                         // 20-DWORD bank stride -> conflict-free ds_load_b128)

__global__ __launch_bounds__(128)
void gemm_xu(const float* __restrict__ x, const __bf16* __restrict__ UT,
             const float* __restrict__ bias, float* __restrict__ xu) {
    __shared__ __bf16 Bs[2][128 * BCOL_STRIDE];   // 2 x 10KB ping-pong

    const int tid  = threadIdx.x;
    const int lane = tid & 31;
    const int wave = tid >> 5;
    const int m0   = blockIdx.y * 128 + wave * 32; // row-tile base in B*T
    const int nblk = blockIdx.x * 128;             // col-tile base in 4H
    const int lm = lane & 15;
    const int lh = lane >> 4;

    // Issue one 32-K B-tile fill into buffer `buf` (4 async b128 per thread,
    // 512 x 16B chunks total). ASYNCcnt += 4 per wave.
    auto issue_fill = [&](int buf, int kk) {
#pragma unroll
        for (int j = 0; j < 4; ++j) {
            int c  = j * 128 + tid;     // chunk id 0..511
            int n  = c >> 2;            // column 0..127
            int kl = (c & 3) * 8;       // K offset within slice (x8 bf16)
            const __bf16* src = UT + (size_t)(nblk + n) * H_SZ + kk + kl;
            // Low 32 bits of a generic LDS pointer == wave-relative LDS addr.
            unsigned dst = (unsigned)(size_t)&Bs[buf][n * BCOL_STRIDE + kl];
            asm volatile("global_load_async_to_lds_b128 %0, %1, off"
                         :: "v"(dst), "v"(src) : "memory");
        }
    };

    v8f acc[2][8] = {};
    const float* xrow0 = x + (size_t)(m0 + lm) * H_SZ;        // A frag 0 rows
    const float* xrow1 = x + (size_t)(m0 + 16 + lm) * H_SZ;   // A frag 1 rows

    issue_fill(0, 0);                       // prologue: buffer 0 in flight

    for (int kk = 0; kk < H_SZ; kk += 32) {
        const int buf = (kk >> 5) & 1;
        if (kk + 32 < H_SZ) {
            issue_fill(buf ^ 1, kk + 32);   // 8 outstanding: 4 old + 4 new
            asm volatile("s_wait_asynccnt 0x4" ::: "memory");  // old 4 done
        } else {
            asm volatile("s_wait_asynccnt 0x0" ::: "memory");
        }
        __syncthreads();                    // B-tile visible to all waves

        // Two A fragments (16x32 bf16 layout): K = ka..ka+15.
        int ka = kk + lh * 16;
        v16bf a0, a1;
        {
            v4f p0 = *(const v4f*)(xrow0 + ka);
            v4f p1 = *(const v4f*)(xrow0 + ka + 4);
            v4f p2 = *(const v4f*)(xrow0 + ka + 8);
            v4f p3 = *(const v4f*)(xrow0 + ka + 12);
            v4f q0 = *(const v4f*)(xrow1 + ka);
            v4f q1 = *(const v4f*)(xrow1 + ka + 4);
            v4f q2 = *(const v4f*)(xrow1 + ka + 8);
            v4f q3 = *(const v4f*)(xrow1 + ka + 12);
#pragma unroll
            for (int j = 0; j < 4; ++j) {
                a0[j]      = (__bf16)p0[j];
                a0[4 + j]  = (__bf16)p1[j];
                a0[8 + j]  = (__bf16)p2[j];
                a0[12 + j] = (__bf16)p3[j];
                a1[j]      = (__bf16)q0[j];
                a1[4 + j]  = (__bf16)q1[j];
                a1[8 + j]  = (__bf16)q2[j];
                a1[12 + j] = (__bf16)q3[j];
            }
        }

#pragma unroll
        for (int s = 0; s < 8; ++s) {
            const __bf16* bp = &Bs[buf][(s * 16 + lm) * BCOL_STRIDE + lh * 16];
            v16bf b = *(const v16bf*)bp;    // 2 x ds_load_b128, K ascending
            acc[0][s] = __builtin_amdgcn_wmma_f32_16x16x32_bf16(
                false, a0, false, b, (short)0, acc[0][s], false, false);
            acc[1][s] = __builtin_amdgcn_wmma_f32_16x16x32_bf16(
                false, a1, false, b, (short)0, acc[1][s], false, false);
        }
        __syncthreads();                    // done reading before refill
    }

    // Epilogue: add bias, remap row (b*T+t) -> (t*B+b) so each timestep's
    // xu slice is contiguous for the scan kernel.
#pragma unroll
    for (int v = 0; v < 2; ++v) {
#pragma unroll
        for (int s = 0; s < 8; ++s) {
            int n = nblk + s * 16 + lm;
            float bv = bias[n];
#pragma unroll
            for (int r = 0; r < 8; ++r) {
                int mm = m0 + v * 16 + lh * 8 + r;  // global row in B*T
                int b  = mm >> 9;                   // / T
                int t  = mm & (T_SZ - 1);           // % T
                xu[(size_t)(t * B_SZ + b) * G_SZ + n] = acc[v][s][r] + bv;
            }
        }
    }
}

// ---------------------------------------------------------------------------
// One LSTM timestep.  gates = xu_t + c @ W  (only i, f, o need the GEMM; g is
// sigmoid of the xu slice alone).  Grid: 32 blocks (one 16-wide H chunk each),
// 4 waves per block covering M = B = 64 rows.  Each wave: 3 WMMA accumulators
// (i/f/o gate columns), K-loop 16 iters of v_wmma_f32_16x16x32_bf16.
// WT stays in L2 across all 512 steps; unroll lets loads run ahead of the
// dependent WMMA chain.
// ---------------------------------------------------------------------------
__global__ __launch_bounds__(128)
void lstm_step(const __bf16* __restrict__ cb_cur, const float* __restrict__ cf_cur,
               const __bf16* __restrict__ WT, const float* __restrict__ xu,
               float* __restrict__ cf_nxt, __bf16* __restrict__ cb_nxt,
               float* __restrict__ hid, int t) {
    const int lane = threadIdx.x & 31;
    const int wave = threadIdx.x >> 5;
    const int m0 = wave * 16;           // batch-row tile
    const int n0 = blockIdx.x * 16;     // H-column tile
    const int lm = lane & 15;
    const int lh = lane >> 4;

    v8f acc_i = {}, acc_f = {}, acc_o = {};

    const __bf16* crow = cb_cur + (size_t)(m0 + lm) * H_SZ;
    const int n = n0 + lm;

#pragma unroll 4
    for (int kk = 0; kk < H_SZ; kk += 32) {
        int ka = kk + lh * 16;
        v16bf a;
        *((v8bf*)&a)     = *(const v8bf*)(crow + ka);        // K ka..ka+7
        *((v8bf*)&a + 1) = *(const v8bf*)(crow + ka + 8);    // K ka+8..ka+15
        int ksb = kk + lh * 16;
        v16bf bi = *(const v16bf*)(WT + (size_t)(0 * H_SZ + n) * H_SZ + ksb);
        v16bf bf = *(const v16bf*)(WT + (size_t)(1 * H_SZ + n) * H_SZ + ksb);
        v16bf bo = *(const v16bf*)(WT + (size_t)(3 * H_SZ + n) * H_SZ + ksb);
        acc_i = __builtin_amdgcn_wmma_f32_16x16x32_bf16(false, a, false, bi,
                                                        (short)0, acc_i, false, false);
        acc_f = __builtin_amdgcn_wmma_f32_16x16x32_bf16(false, a, false, bf,
                                                        (short)0, acc_f, false, false);
        acc_o = __builtin_amdgcn_wmma_f32_16x16x32_bf16(false, a, false, bo,
                                                        (short)0, acc_o, false, false);
    }

    const float* xub = xu + (size_t)t * B_SZ * G_SZ;
#pragma unroll
    for (int r = 0; r < 8; ++r) {
        int m = m0 + lh * 8 + r;                 // batch index
        const float* xr = xub + (size_t)m * G_SZ;
        float gi = acc_i[r] + xr[0 * H_SZ + n];
        float gf = acc_f[r] + xr[1 * H_SZ + n];
        float gg =            xr[2 * H_SZ + n];  // g uses xu only
        float go = acc_o[r] + xr[3 * H_SZ + n];
        float i_ = 1.0f / (1.0f + __expf(-gi));
        float f_ = 1.0f / (1.0f + __expf(-gf));
        float g_ = 1.0f / (1.0f + __expf(-gg));
        float o_ = 1.0f / (1.0f + __expf(-go));
        float c_old = cf_cur[(size_t)m * H_SZ + n];
        float c_new = f_ * c_old + i_ * g_;
        float h = tanhf(o_ * c_new);
        cf_nxt[(size_t)m * H_SZ + n] = c_new;
        cb_nxt[(size_t)m * H_SZ + n] = (__bf16)c_new;
        hid[((size_t)m * T_SZ + t) * H_SZ + n] = h;  // hidden_seq[b, t, h]
    }
}

// Copy h_last (last timestep of hidden_seq) and c_last into d_out tail.
__global__ void finalize(const float* __restrict__ cf, const float* __restrict__ hid,
                         float* __restrict__ hlast, float* __restrict__ clast) {
    int idx = blockIdx.x * blockDim.x + threadIdx.x;   // over B*H
    int b = idx >> 9;
    int h = idx & (H_SZ - 1);
    hlast[idx] = hid[((size_t)b * T_SZ + (T_SZ - 1)) * H_SZ + h];
    clast[idx] = cf[idx];
}

extern "C" void kernel_launch(void* const* d_in, const int* in_sizes, int n_in,
                              void* d_out, int out_size, void* d_ws, size_t ws_size,
                              hipStream_t stream) {
    (void)in_sizes; (void)n_in; (void)out_size; (void)ws_size;
    const float* x    = (const float*)d_in[0];   // (B, T, H)
    // d_in[1] = h0: not needed (h is not recurrent in this peephole variant)
    const float* c0   = (const float*)d_in[2];   // (B, H)
    const float* U    = (const float*)d_in[3];   // (H, 4H)
    const float* W    = (const float*)d_in[4];   // (H, 4H)
    const float* bias = (const float*)d_in[5];   // (4H,)
    float* out = (float*)d_out;                  // hidden_seq | h_last | c_last

    // Workspace layout
    char* ws = (char*)d_ws;
    size_t off = 0;
    float*  xu  = (float*)(ws + off);  off += (size_t)T_SZ * B_SZ * G_SZ * sizeof(float);   // 256 MB
    __bf16* UT  = (__bf16*)(ws + off); off += (size_t)G_SZ * H_SZ * sizeof(__bf16);         // 2 MB
    __bf16* WT  = (__bf16*)(ws + off); off += (size_t)G_SZ * H_SZ * sizeof(__bf16);         // 2 MB
    float*  cf0 = (float*)(ws + off);  off += (size_t)B_SZ * H_SZ * sizeof(float);
    float*  cf1 = (float*)(ws + off);  off += (size_t)B_SZ * H_SZ * sizeof(float);
    __bf16* cb0 = (__bf16*)(ws + off); off += (size_t)B_SZ * H_SZ * sizeof(__bf16);
    __bf16* cb1 = (__bf16*)(ws + off); off += (size_t)B_SZ * H_SZ * sizeof(__bf16);
    float*  cfb[2] = {cf0, cf1};
    __bf16* cbb[2] = {cb0, cb1};

    // 1) Pack weights: bf16 transposed [4H][H] for contiguous B-fragment loads
    pack_transpose_bf16<<<(G_SZ * H_SZ) / 256, 256, 0, stream>>>(U, UT);
    pack_transpose_bf16<<<(G_SZ * H_SZ) / 256, 256, 0, stream>>>(W, WT);

    // 2) Initialize cell state ping-pong buffers
    init_c<<<(B_SZ * H_SZ) / 256, 256, 0, stream>>>(c0, cfb[0], cbb[0]);

    // 3) Big parallel GEMM: xu = x @ U + bias (remapped to (t, b, 4H) order)
    gemm_xu<<<dim3(G_SZ / 128, (B_SZ * T_SZ) / 128), 128, 0, stream>>>(x, UT, bias, xu);

    // 4) Sequential scan: one launch per timestep (dependency via launch order)
    for (int t = 0; t < T_SZ; ++t) {
        int cur = t & 1;
        lstm_step<<<H_SZ / 16, 128, 0, stream>>>(cbb[cur], cfb[cur], WT, xu,
                                                 cfb[1 - cur], cbb[1 - cur], out, t);
    }
    // After step T-1 (odd index 511), final state lives in buffer 0.

    // 5) h_last, c_last
    finalize<<<(B_SZ * H_SZ) / 256, 256, 0, stream>>>(
        cfb[0], out,
        out + (size_t)B_SZ * T_SZ * H_SZ,
        out + (size_t)B_SZ * T_SZ * H_SZ + (size_t)B_SZ * H_SZ);
}
